// Mamba_SSM_12824772346535
// MI455X (gfx1250) — matrix operand order repaired and checked by reading, not compiled, not run on hardware
//
#include <hip/hip_runtime.h>
#include <hip/hip_bf16.h>

#define B_  2
#define L_  1024
#define D_  768
#define E_  1536
#define N_  16
#define R_  48
#define K_  4
#define NL_ 4
#define NC_ 5
#define EPS_ 1e-5f
#define M_  (B_*L_)          // 2048 rows
#define XW_ (R_ + 2*N_)      // 80
#define RP_ 64               // dt_w K padded to 64 (multiple of 32)

typedef __attribute__((ext_vector_type(16))) _Float16 v16h;
typedef __attribute__((ext_vector_type(8)))  _Float16 v8h;
typedef __attribute__((ext_vector_type(4)))  _Float16 v4h;
typedef __attribute__((ext_vector_type(8)))  float    v8f;

__device__ __forceinline__ float sigmoidf_(float x) { return 1.f / (1.f + __expf(-x)); }

// ---------------- input projection: h = x @ proj_w^T + proj_b ----------------
__global__ void k_proj(const float* __restrict__ x, const float* __restrict__ w,
                       const float* __restrict__ b, float* __restrict__ h)
{
    int idx = blockIdx.x * blockDim.x + threadIdx.x;
    if (idx >= M_ * D_) return;
    int row = idx / D_, d = idx % D_;
    const float* xr = x + (size_t)row * 12;
    const float* wr = w + (size_t)d * 12;
    float acc = b[d];
#pragma unroll
    for (int j = 0; j < 12; ++j) acc += xr[j] * wr[j];
    h[idx] = acc;
}

// ---------------- f32 -> f16 weight conversion (n multiple of 4) ----------------
__global__ void k_cvt16(const float* __restrict__ in, _Float16* __restrict__ out, int n4)
{
    int i = blockIdx.x * blockDim.x + threadIdx.x;
    if (i >= n4) return;
    float4 v = ((const float4*)in)[i];
    v4h o; o[0] = (_Float16)v.x; o[1] = (_Float16)v.y; o[2] = (_Float16)v.z; o[3] = (_Float16)v.w;
    ((v4h*)out)[i] = o;
}

// ---------------- dt_w (E,48) -> f16 zero-padded (E,64) ----------------
__global__ void k_pad_dtw(const float* __restrict__ dtw, _Float16* __restrict__ out)
{
    int idx = blockIdx.x * blockDim.x + threadIdx.x;
    if (idx >= E_ * RP_) return;
    int e = idx >> 6, k = idx & (RP_ - 1);
    out[idx] = (k < R_) ? (_Float16)dtw[e * R_ + k] : (_Float16)0.f;
}

// ---------------- RMSNorm over last dim; writes f32 + f16 ----------------
__global__ void k_rmsnorm(const float* __restrict__ h, const float* __restrict__ w,
                          float* __restrict__ out, _Float16* __restrict__ out16)
{
    __shared__ float red[256];
    int row = blockIdx.x;
    const float* hr = h + (size_t)row * D_;
    float s = 0.f;
    for (int d = threadIdx.x; d < D_; d += 256) { float v = hr[d]; s += v * v; }
    red[threadIdx.x] = s;
    __syncthreads();
    for (int off = 128; off > 0; off >>= 1) {
        if (threadIdx.x < off) red[threadIdx.x] += red[threadIdx.x + off];
        __syncthreads();
    }
    float rinv = rsqrtf(red[0] / (float)D_ + EPS_);
    for (int d = threadIdx.x; d < D_; d += 256) {
        float v = hr[d] * rinv * w[d];
        out[(size_t)row * D_ + d] = v;
        out16[(size_t)row * D_ + d] = (_Float16)v;
    }
}

// ---------------- generic WMMA GEMM (f16 in, f32 out): C = A @ W^T ----------------
// A,W pre-converted to f16; K multiple of 32; NT = active 16-col sub-tiles.
// Per K-step: 2x b128 A loads + NT*2x b128 B loads + NT v_wmma (no cvt in loop).
// EPI: 0 = store, 1 = +bias softplus, 2 = +residual.  W16: also store f16 copy.
template<int EPI, int NT, bool W16>
__global__ void k_gemm_wmma(const _Float16* __restrict__ A, int lda,
                            const _Float16* __restrict__ W, int ldw,
                            float* __restrict__ C, int ldc,
                            _Float16* __restrict__ C16,
                            const float* __restrict__ bias,
                            const float* __restrict__ Res, int ldr,
                            int nbase, int K)
{
    const int lane   = threadIdx.x;       // 0..31
    const int half   = lane >> 4;         // 0 or 1
    const int lane16 = lane & 15;
    const int mt = blockIdx.y;
    const int n0 = nbase + blockIdx.x * (NT * 16);

    const _Float16* arow = A + (size_t)(mt * 16 + lane16) * lda;

    const _Float16* wrow[NT];
#pragma unroll
    for (int j = 0; j < NT; ++j)
        wrow[j] = W + (size_t)(n0 + j * 16 + lane16) * ldw;

    v8f acc[NT];
#pragma unroll
    for (int j = 0; j < NT; ++j) acc[j] = v8f{};

    for (int k0 = 0; k0 < K; k0 += 32) {
        // A fragment (16x32 f16): halfs[0..7] = K[k0+16*half ..], halfs[8..15] = K[k0+16*half+8 ..]
        const v8h lo = *(const v8h*)(arow + k0 + 16 * half);
        const v8h hi = *(const v8h*)(arow + k0 + 16 * half + 8);
        v16h a;
#pragma unroll
        for (int i = 0; i < 8; ++i) { a[i] = lo[i]; a[8 + i] = hi[i]; }

        // B fragment: lane holds col lane16; halfs[h] = W[col][k0 + 16*half + h]
        const int kb = k0 + 16 * half;
#pragma unroll
        for (int j = 0; j < NT; ++j) {
            const v16h bf = *(const v16h*)(wrow[j] + kb);
            acc[j] = __builtin_amdgcn_wmma_f32_16x16x32_f16(false, a, false, bf,
                                                            (short)0, acc[j], false, false);
        }
    }

    // C/D layout: float v of accumulator -> row = v + 8*half, col = lane16
#pragma unroll
    for (int j = 0; j < NT; ++j) {
        int n = n0 + j * 16 + lane16;
#pragma unroll
        for (int v = 0; v < 8; ++v) {
            int row = mt * 16 + v + 8 * half;
            float val = acc[j][v];
            if (EPI == 1) { val += bias[n]; val = (val > 20.f) ? val : log1pf(__expf(val)); }
            if (EPI == 2) { val += Res[(size_t)row * ldr + n]; }
            C[(size_t)row * ldc + n] = val;
            if (W16) C16[(size_t)row * ldc + n] = (_Float16)val;
        }
    }
}

// ---------------- causal depthwise conv (K=4) + SiLU; writes f32 + f16 ----------------
__global__ void k_conv_silu(const float* __restrict__ proj, const float* __restrict__ cw,
                            const float* __restrict__ cb, float* __restrict__ u,
                            _Float16* __restrict__ u16)
{
    int idx = blockIdx.x * blockDim.x + threadIdx.x;
    if (idx >= B_ * L_ * E_) return;
    int e = idx % E_;
    int l = (idx / E_) % L_;
    int b = idx / (E_ * L_);
    float acc = cb[e];
#pragma unroll
    for (int k = 0; k < K_; ++k) {
        int lt = l - (K_ - 1) + k;
        float v = (lt >= 0) ? proj[(size_t)(b * L_ + lt) * (2 * E_) + e] : 0.f;
        acc += cw[e * K_ + k] * v;
    }
    float s = acc * sigmoidf_(acc);   // silu
    u[idx] = s;
    u16[idx] = (_Float16)s;
}

// ---------------- selective scan, fused gate*silu; writes f16 for out-GEMM ----------------
__global__ void k_scan(const float* __restrict__ u, const float* __restrict__ dt,
                       const float* __restrict__ A_log, const float* __restrict__ xdbc,
                       const float* __restrict__ Dp, const float* __restrict__ proj,
                       _Float16* __restrict__ ys16)
{
    __shared__ float sB[N_];
    __shared__ float sC[N_];
    int e = blockIdx.x * blockDim.x + threadIdx.x;
    int b = blockIdx.y;

    float Am[N_];
#pragma unroll
    for (int n = 0; n < N_; ++n) Am[n] = -__expf(A_log[(size_t)e * N_ + n]);
    float Dv = Dp[e];
    float hs[N_];
#pragma unroll
    for (int n = 0; n < N_; ++n) hs[n] = 0.f;

    for (int t = 0; t < L_; ++t) {
        size_t rowx = (size_t)(b * L_ + t);
        if (threadIdx.x < 32)    // lanes 0..15 -> B (48..63), 16..31 -> C (64..79)
            ((threadIdx.x < 16) ? sB[threadIdx.x] : sC[threadIdx.x - 16]) =
                xdbc[rowx * XW_ + R_ + threadIdx.x];
        __syncthreads();

        float dtv = dt[rowx * E_ + e];
        float uv  = u[rowx * E_ + e];
        float y = 0.f;
#pragma unroll
        for (int n = 0; n < N_; ++n) {
            float dA = __expf(dtv * Am[n]);
            hs[n] = hs[n] * dA + dtv * sB[n] * uv;
            y += hs[n] * sC[n];
        }
        y += uv * Dv;
        float g = proj[rowx * (2 * E_) + E_ + e];
        y *= g * sigmoidf_(g);
        ys16[rowx * E_ + e] = (_Float16)y;
        __syncthreads();
    }
}

// ---------------- mean pool over L ----------------
__global__ void k_pool(const float* __restrict__ hn, float* __restrict__ pooled)
{
    int idx = blockIdx.x * blockDim.x + threadIdx.x;
    if (idx >= B_ * D_) return;
    int d = idx % D_, b = idx / D_;
    float s = 0.f;
    for (int l = 0; l < L_; ++l) s += hn[(size_t)(b * L_ + l) * D_ + d];
    pooled[idx] = s / (float)L_;
}

// ---------------- classifier ----------------
__global__ void k_cls(const float* __restrict__ pooled, const float* __restrict__ cw,
                      const float* __restrict__ cb, float* __restrict__ out)
{
    int t = threadIdx.x;
    if (t >= B_ * NC_) return;
    int b = t / NC_, c = t % NC_;
    float s = cb[c];
    const float* p = pooled + (size_t)b * D_;
    const float* w = cw + (size_t)c * D_;
    for (int d = 0; d < D_; ++d) s += p[d] * w[d];
    out[t] = s;
}

extern "C" void kernel_launch(void* const* d_in, const int* in_sizes, int n_in,
                              void* d_out, int out_size, void* d_ws, size_t ws_size,
                              hipStream_t stream)
{
    const float* x       = (const float*)d_in[0];
    const float* proj_w  = (const float*)d_in[1];
    const float* proj_b  = (const float*)d_in[2];
    const float* norm_w  = (const float*)d_in[3];
    const float* in_w    = (const float*)d_in[4];
    const float* conv_w  = (const float*)d_in[5];
    const float* conv_b  = (const float*)d_in[6];
    const float* xproj_w = (const float*)d_in[7];
    const float* dt_w    = (const float*)d_in[8];
    const float* dt_b    = (const float*)d_in[9];
    const float* A_log   = (const float*)d_in[10];
    const float* D_par   = (const float*)d_in[11];
    const float* out_w   = (const float*)d_in[12];
    const float* fnorm_w = (const float*)d_in[13];
    const float* cls_w   = (const float*)d_in[14];
    const float* cls_b   = (const float*)d_in[15];
    float* out = (float*)d_out;

    // ---- f32 region ----
    float* ws = (float*)d_ws;
    size_t off = 0;
    float* h      = ws + off; off += (size_t)M_ * D_;
    float* proj   = ws + off; off += (size_t)M_ * 2 * E_;
    float* u      = ws + off; off += (size_t)M_ * E_;
    float* xdbc   = ws + off; off += (size_t)M_ * XW_;
    float* dtb    = ws + off; off += (size_t)M_ * E_;
    float* hn     = ws + off; off += (size_t)M_ * D_;
    float* pooled = ws + off; off += (size_t)B_ * D_;
    // ---- f16 region (32B-aligned: preceding counts are multiples of 16) ----
    _Float16* f16base = (_Float16*)(ws + off);
    size_t off16 = 0;
    _Float16* hn16   = f16base + off16; off16 += (size_t)M_ * D_;
    _Float16* u16    = f16base + off16; off16 += (size_t)M_ * E_;
    _Float16* xdbc16 = f16base + off16; off16 += (size_t)M_ * XW_;
    _Float16* ys16   = f16base + off16; off16 += (size_t)M_ * E_;
    _Float16* wbuf16 = f16base + off16; off16 += (size_t)2 * E_ * D_;  // largest weight
    _Float16* dtw16  = f16base + off16; off16 += (size_t)E_ * RP_;
    (void)in_sizes; (void)n_in; (void)out_size; (void)ws_size;

    k_proj<<<dim3((M_ * D_ + 255) / 256), 256, 0, stream>>>(x, proj_w, proj_b, h);

    for (int i = 0; i < NL_; ++i) {
        k_rmsnorm<<<M_, 256, 0, stream>>>(h, norm_w + (size_t)i * D_, hn, hn16);

        // proj = hn @ in_w[i]^T   (N=3072, K=768)
        k_cvt16<<<dim3((2 * E_ * D_ / 4 + 255) / 256), 256, 0, stream>>>(
            in_w + (size_t)i * 2 * E_ * D_, wbuf16, 2 * E_ * D_ / 4);
        k_gemm_wmma<0, 4, false><<<dim3((2 * E_) / 64, M_ / 16), 32, 0, stream>>>(
            hn16, D_, wbuf16, D_, proj, 2 * E_, nullptr, nullptr, nullptr, 0, 0, D_);

        k_conv_silu<<<dim3((B_ * L_ * E_ + 255) / 256), 256, 0, stream>>>(
            proj, conv_w + (size_t)i * E_ * K_, conv_b + (size_t)i * E_, u, u16);

        // xdbc = u @ xproj_w[i]^T  (N=80, K=1536): 64 cols NT=4 + 16-col tail NT=1
        k_cvt16<<<dim3((XW_ * E_ / 4 + 255) / 256), 256, 0, stream>>>(
            xproj_w + (size_t)i * XW_ * E_, wbuf16, XW_ * E_ / 4);
        k_gemm_wmma<0, 4, true><<<dim3(1, M_ / 16), 32, 0, stream>>>(
            u16, E_, wbuf16, E_, xdbc, XW_, xdbc16, nullptr, nullptr, 0, 0, E_);
        k_gemm_wmma<0, 1, true><<<dim3(1, M_ / 16), 32, 0, stream>>>(
            u16, E_, wbuf16, E_, xdbc, XW_, xdbc16, nullptr, nullptr, 0, 64, E_);

        // dt = softplus(xdbc16[:, 0:64] @ dtw16^T + dt_b[i]) — padded cols hit zero weights
        k_pad_dtw<<<dim3((E_ * RP_ + 255) / 256), 256, 0, stream>>>(
            dt_w + (size_t)i * E_ * R_, dtw16);
        k_gemm_wmma<1, 4, false><<<dim3(E_ / 64, M_ / 16), 32, 0, stream>>>(
            xdbc16, XW_, dtw16, RP_, dtb, E_, nullptr,
            dt_b + (size_t)i * E_, nullptr, 0, 0, RP_);

        k_scan<<<dim3(E_ / 256, B_), 256, 0, stream>>>(
            u, dtb, A_log + (size_t)i * E_ * N_, xdbc, D_par + (size_t)i * E_, proj, ys16);

        // h = h + ys @ out_w[i]^T  (N=768, K=1536, residual, in-place)
        k_cvt16<<<dim3((D_ * E_ / 4 + 255) / 256), 256, 0, stream>>>(
            out_w + (size_t)i * D_ * E_, wbuf16, D_ * E_ / 4);
        k_gemm_wmma<2, 4, false><<<dim3(D_ / 64, M_ / 16), 32, 0, stream>>>(
            ys16, E_, wbuf16, E_, h, D_, nullptr, nullptr, h, D_, 0, E_);
    }

    k_rmsnorm<<<M_, 256, 0, stream>>>(h, fnorm_w, hn, hn16);
    k_pool<<<dim3((B_ * D_ + 255) / 256), 256, 0, stream>>>(hn, pooled);
    k_cls<<<1, 64, 0, stream>>>(pooled, cls_w, cls_b, out);
}
